// PPN_68547678044521
// MI455X (gfx1250) — matrix operand order repaired and checked by reading, not compiled, not run on hardware
//
#include <hip/hip_runtime.h>
#include <hip/hip_bf16.h>

// ---------------------------------------------------------------------------
// PPN forward for MI455X (gfx1250, wave32, WMMA).
// Dominant cost (the 20-iter convLSTM, ~510 GFLOP of the ~0.55 TFLOP total)
// is folded into a single 3x3 conv with 256 outputs and executed with
// v_wmma_f32_16x16x32_f16; h/c/gates live in LDS (163 KB/WGP), weights are
// pre-swizzled into per-lane B-fragment layout and stay L2-resident.
// GEMM inner loop is software-pipelined (A frags double-buffered vs WMMA,
// B frags double-buffered across K-chunks).  The K loop is explicitly kept
// ROLLED: round-2 showed that full unroll makes the scheduler hoist A loads
// until it spills fragments to scratch (scratch_load_b128 feeding WMMA).
// ---------------------------------------------------------------------------

typedef __attribute__((ext_vector_type(16))) _Float16 v16h;
typedef __attribute__((ext_vector_type(8)))  float    v8f;
typedef __attribute__((ext_vector_type(4)))  unsigned u32x4;

#define kB    512
#define kW    13
#define NPIX  169      // 13*13
#define HP    15       // 13 + halo
#define CAPS  8
#define NGATE 256
#define KCH   18       // 576 / 32
#define MT    11       // ceil(169/16)
#define NT    16       // 256 / 16
#define NITER 20

static __device__ __forceinline__ float sigmoidf_(float x) { return 1.f / (1.f + __expf(-x)); }
static __device__ __forceinline__ float leaky_(float x)    { return x > 0.f ? x : 0.01f * x; }

// ---------------------------------------------------------------------------
// Attention: per-pixel MLP 4->16->16->32->32->64->1, softmax over 169 pixels,
// times softmax-over-CAP head from the state MLP.  One block per (batch,which).
// ---------------------------------------------------------------------------
struct AttnW {
  const float *w0, *b0, *w1, *b1, *w2, *b2, *w3, *b3, *w4, *b4, *w5, *b5;
  const float *mw1, *mb1, *mw2, *mb2;
};

__global__ void __launch_bounds__(192)
attn_kernel(const float* __restrict__ goal, const float* __restrict__ cur,
            AttnW p, float* __restrict__ gatt, float* __restrict__ satt)
{
  __shared__ float logits[NPIX];
  __shared__ float a3[CAPS];
  __shared__ float red2[2];

  const int b     = blockIdx.x;
  const int which = blockIdx.y;               // 0 = goal, 1 = cur
  const float* st = which ? cur : goal;
  float* outp     = which ? satt : gatt;
  const int tid   = threadIdx.x;

  const float sx = st[b * 2 + 0], sy = st[b * 2 + 1];

  if (tid < NPIX) {
    const int py = tid / kW, px = tid - py * kW;
    const float x0 = sx, x1 = sy, x2 = (float)px, x3 = (float)py;
    float t0[16];
#pragma unroll
    for (int o = 0; o < 16; ++o)
      t0[o] = fmaxf(p.b0[o] + p.w0[o*4+0]*x0 + p.w0[o*4+1]*x1 + p.w0[o*4+2]*x2 + p.w0[o*4+3]*x3, 0.f);
    float t1[16];
#pragma unroll
    for (int o = 0; o < 16; ++o) {
      float a = p.b1[o];
#pragma unroll
      for (int i = 0; i < 16; ++i) a += p.w1[o*16+i] * t0[i];
      t1[o] = fmaxf(a, 0.f);
    }
    float t2[32];
#pragma unroll
    for (int o = 0; o < 32; ++o) {
      float a = p.b2[o];
#pragma unroll
      for (int i = 0; i < 16; ++i) a += p.w2[o*16+i] * t1[i];
      t2[o] = fmaxf(a, 0.f);
    }
    float t3[32];
#pragma unroll
    for (int o = 0; o < 32; ++o) {
      float a = p.b3[o];
#pragma unroll
      for (int i = 0; i < 32; ++i) a += p.w3[o*32+i] * t2[i];
      t3[o] = fmaxf(a, 0.f);
    }
    float lg = p.b5[0];
#pragma unroll
    for (int o = 0; o < 64; ++o) {
      float a = p.b4[o];
#pragma unroll
      for (int i = 0; i < 32; ++i) a += p.w4[o*32+i] * t3[i];
      lg += p.w5[o] * fmaxf(a, 0.f);
    }
    logits[tid] = lg;
  }
  __syncthreads();

  if (tid == 0) {
    float mx = logits[0];
    for (int i = 1; i < NPIX; ++i) mx = fmaxf(mx, logits[i]);
    float s = 0.f;
    for (int i = 0; i < NPIX; ++i) s += __expf(logits[i] - mx);
    red2[0] = mx; red2[1] = 1.f / s;
  } else if (tid == 32) {                      // head MLP on another wave
    float sv[64];
    for (int o = 0; o < 64; ++o)
      sv[o] = fmaxf(p.mw1[o*2+0]*sx + p.mw1[o*2+1]*sy + p.mb1[o], 0.f);
    float l8[CAPS]; float mx = -1e30f;
    for (int o = 0; o < CAPS; ++o) {
      float a = p.mb2[o];
      for (int k = 0; k < 64; ++k) a += p.mw2[o*64+k] * sv[k];
      l8[o] = a; mx = fmaxf(mx, a);
    }
    float s = 0.f;
    for (int o = 0; o < CAPS; ++o) { l8[o] = __expf(l8[o]-mx); s += l8[o]; }
    for (int o = 0; o < CAPS; ++o) a3[o] = l8[o] / s;
  }
  __syncthreads();

  const float mx = red2[0], inv = red2[1];
  for (int idx = tid; idx < CAPS * NPIX; idx += 192) {
    const int cap = idx / NPIX, pp = idx - cap * NPIX;
    outp[((size_t)b * CAPS + cap) * NPIX + pp] = __expf(logits[pp] - mx) * inv * a3[cap];
  }
}

// ---------------------------------------------------------------------------
// Prep: fold W_ih@conv_w (+W_hh on center tap) into a 576x256 f16 matrix
// stored in exact per-lane WMMA B-fragment layout [kchunk][ntile][lane][8 dw];
// fold b_ih + b_hh + W_ih@conv_b into a 256-float gate bias.
// ---------------------------------------------------------------------------
__global__ void __launch_bounds__(256)
prep_kernel(const float* __restrict__ conv_w,     // [64][64][3][3]
            const float* __restrict__ w_ih,       // [256][64]
            const float* __restrict__ w_hh,       // [256][64]
            const float* __restrict__ b_ih,       // [256]
            const float* __restrict__ b_hh,       // [256]
            const float* __restrict__ conv_b,     // [64]
            unsigned* __restrict__ wgfrag,        // [18][16][32][8] dwords
            float* __restrict__ biasg)            // [256]
{
  const int blk = blockIdx.x;                     // 18*16 blocks
  const int kc = blk >> 4, nt = blk & 15;
  const int tid = threadIdx.x;
  const int lane = tid >> 3;                      // 0..31
  const int r = tid & 7;                          // dword within fragment
  const int o = nt * 16 + (lane & 15);            // gate channel (column N)
  const int kbase = kc * 32 + ((lane < 16) ? 0 : 8) + (r & 3) * 2 + (r >> 2) * 16;

  _Float16 hv[2];
#pragma unroll
  for (int h = 0; h < 2; ++h) {
    const int K = kbase + h;                      // K = tap*64 + ci
    const int t = K >> 6, ci = K & 63;
    float acc = (t == 4) ? w_hh[o*64 + ci] : 0.f; // center tap gets W_hh
    for (int m = 0; m < 64; ++m)
      acc += w_ih[o*64 + m] * conv_w[(m*64 + ci)*9 + t];
    hv[h] = (_Float16)acc;
  }
  union { _Float16 f[2]; unsigned u; } pk;
  pk.f[0] = hv[0]; pk.f[1] = hv[1];
  wgfrag[((kc * NT + nt) * 32 + lane) * 8 + r] = pk.u;

  if (blk == 0) {
    float a = b_ih[tid] + b_hh[tid];
    for (int m = 0; m < 64; ++m) a += w_ih[tid*64 + m] * conv_b[m];
    biasg[tid] = a;
  }
}

// ---------------------------------------------------------------------------
// Maze encoder convs (direct, f32 VALU; <3% of FLOPs)
// ---------------------------------------------------------------------------
__global__ void conv1_kernel(const float* __restrict__ maze,     // [B][1][101][101]
                             const float* __restrict__ w,        // [16][1][4][4]
                             const float* __restrict__ bias,     // [16]
                             float* __restrict__ out)            // [B][16][50][50]
{
  int idx = blockIdx.x * 256 + threadIdx.x;
  if (idx >= kB * 16 * 50 * 50) return;
  int ox = idx % 50; int t = idx / 50;
  int oy = t % 50;   t /= 50;
  int oc = t & 15;   int b = t >> 4;
  float acc = bias[oc];
  for (int ky = 0; ky < 4; ++ky) {
    int iy = oy * 2 - 1 + ky; if (iy < 0 || iy >= 101) continue;
    for (int kx = 0; kx < 4; ++kx) {
      int ix = ox * 2 - 1 + kx; if (ix < 0 || ix >= 101) continue;
      acc += w[oc * 16 + ky * 4 + kx] * maze[((size_t)b * 101 + iy) * 101 + ix];
    }
  }
  out[idx] = leaky_(acc);
}

__global__ void conv2_kernel(const float* __restrict__ in,       // [B][16][50][50]
                             const float* __restrict__ w,        // [32][16][3][3]
                             const float* __restrict__ cb,
                             const float* __restrict__ g, const float* __restrict__ bb,
                             const float* __restrict__ m, const float* __restrict__ v,
                             float* __restrict__ out)            // [B][32][25][25]
{
  int idx = blockIdx.x * 256 + threadIdx.x;
  if (idx >= kB * 32 * 25 * 25) return;
  int ox = idx % 25; int t = idx / 25;
  int oy = t % 25;   t /= 25;
  int oc = t & 31;   int b = t >> 5;
  float acc = cb[oc];
  for (int ic = 0; ic < 16; ++ic)
    for (int ky = 0; ky < 3; ++ky) {
      int iy = oy * 2 - 1 + ky; if (iy < 0 || iy >= 50) continue;
      for (int kx = 0; kx < 3; ++kx) {
        int ix = ox * 2 - 1 + kx; if (ix < 0 || ix >= 50) continue;
        acc += w[((oc * 16 + ic) * 3 + ky) * 3 + kx] * in[(((size_t)b * 16 + ic) * 50 + iy) * 50 + ix];
      }
    }
  acc = (acc - m[oc]) * (g[oc] * rsqrtf(v[oc] + 1e-5f)) + bb[oc];
  out[idx] = leaky_(acc);
}

__global__ void conv3_kernel(const float* __restrict__ in,       // [B][32][25][25]
                             const float* __restrict__ w,        // [64][32][3][3]
                             const float* __restrict__ cb,
                             const float* __restrict__ g, const float* __restrict__ bb,
                             const float* __restrict__ m, const float* __restrict__ v,
                             float* __restrict__ out)            // [B][64][13][13]
{
  int idx = blockIdx.x * 256 + threadIdx.x;
  if (idx >= kB * 64 * NPIX) return;
  int p = idx % NPIX; int t = idx / NPIX;
  int oc = t & 63;    int b = t >> 6;
  int oy = p / kW, ox = p - oy * kW;
  float acc = cb[oc];
  for (int ic = 0; ic < 32; ++ic)
    for (int ky = 0; ky < 3; ++ky) {
      int iy = oy * 2 - 1 + ky; if (iy < 0 || iy >= 25) continue;
      for (int kx = 0; kx < 3; ++kx) {
        int ix = ox * 2 - 1 + kx; if (ix < 0 || ix >= 25) continue;
        acc += w[((oc * 32 + ic) * 3 + ky) * 3 + kx] * in[(((size_t)b * 32 + ic) * 25 + iy) * 25 + ix];
      }
    }
  acc = (acc - m[oc]) * (g[oc] * rsqrtf(v[oc] + 1e-5f)) + bb[oc];
  out[idx] = leaky_(acc);
}

// hid conv: concat(m[64], goal_atten[8]) -> 64, 3x3, no activation
__global__ void hid_kernel(const float* __restrict__ mf,         // [B][64][169]
                           const float* __restrict__ ga,         // [B][8][169]
                           const float* __restrict__ w,          // [64][72][3][3]
                           const float* __restrict__ bias,       // [64]
                           float* __restrict__ out)              // [B][64][169]
{
  int idx = blockIdx.x * 256 + threadIdx.x;
  if (idx >= kB * 64 * NPIX) return;
  int p = idx % NPIX; int t = idx / NPIX;
  int oc = t & 63;    int b = t >> 6;
  int py = p / kW, px = p - py * kW;
  float acc = bias[oc];
  for (int ky = 0; ky < 3; ++ky) {
    int iy = py - 1 + ky; if (iy < 0 || iy >= kW) continue;
    for (int kx = 0; kx < 3; ++kx) {
      int ix = px - 1 + kx; if (ix < 0 || ix >= kW) continue;
      const int wo = ky * 3 + kx, sp = iy * kW + ix;
      for (int ic = 0; ic < 64; ++ic)
        acc += w[((oc * 72 + ic) * 9) + wo] * mf[(((size_t)b * 64 + ic) * NPIX) + sp];
      for (int ic = 0; ic < 8; ++ic)
        acc += w[((oc * 72 + 64 + ic) * 9) + wo] * ga[(((size_t)b * 8 + ic) * NPIX) + sp];
    }
  }
  out[idx] = acc;
}

// h0/c0 convs (64->64, 3x3, linear); writes pixel-major channel-last
__global__ void h0c0_kernel(const float* __restrict__ hl,        // [B][64][169]
                            const float* __restrict__ h0w, const float* __restrict__ h0b,
                            const float* __restrict__ c0w, const float* __restrict__ c0b,
                            _Float16* __restrict__ h0out,        // [B][169][64]
                            float* __restrict__ c0out)           // [B][169][64]
{
  int idx = blockIdx.x * 256 + threadIdx.x;
  if (idx >= kB * 64 * NPIX) return;
  int p = idx % NPIX; int t = idx / NPIX;
  int oc = t & 63;    int b = t >> 6;
  int py = p / kW, px = p - py * kW;
  float ah = h0b[oc], ac = c0b[oc];
  for (int ky = 0; ky < 3; ++ky) {
    int iy = py - 1 + ky; if (iy < 0 || iy >= kW) continue;
    for (int kx = 0; kx < 3; ++kx) {
      int ix = px - 1 + kx; if (ix < 0 || ix >= kW) continue;
      const int wo = ky * 3 + kx, sp = iy * kW + ix;
      const float* in = hl + ((size_t)b * 64) * NPIX + sp;
      for (int ic = 0; ic < 64; ++ic) {
        float x = in[ic * NPIX];
        ah += h0w[(oc * 64 + ic) * 9 + wo] * x;
        ac += c0w[(oc * 64 + ic) * 9 + wo] * x;
      }
    }
  }
  h0out[((size_t)b * NPIX + p) * 64 + oc] = (_Float16)ah;
  c0out[((size_t)b * NPIX + p) * 64 + oc] = ac;
}

// ---------------------------------------------------------------------------
// convLSTM megakernel: one workgroup (8 waves) per batch element; 20 iters of
// [GEMM(169x256x576) via v_wmma_f32_16x16x32_f16 -> pointwise LSTM update]
// entirely in LDS.  Software-pipelined with the K loop kept rolled so the
// register allocator stays inside the 256-VGPR window (no scratch spills).
// ---------------------------------------------------------------------------
union Frag { u32x4 u[2]; v16h h; };

__global__ void __launch_bounds__(256)
lstm_kernel(const _Float16* __restrict__ h0,      // [B][169][64]
            const float* __restrict__ c0,         // [B][169][64]
            const u32x4* __restrict__ wg,         // [18][16][32][2] uint4
            const float* __restrict__ biasg,      // [256]
            _Float16* __restrict__ hout)          // [B][169][64]
{
  __shared__ __align__(16) _Float16 h_lds[HP * HP * 64];   // 28800 B (zero halo)
  __shared__ float                  c_lds[NPIX * 64];      // 43264 B
  __shared__ __align__(16) _Float16 g_lds[176 * NGATE];    // 90112 B
  __shared__ float                  bias_s[NGATE];         //  1024 B

  const int tid  = threadIdx.x;
  const int b    = blockIdx.x;
  const int lane = tid & 31;
  const int wave = tid >> 5;

  for (int i = tid; i < HP * HP * 64; i += 256) h_lds[i] = (_Float16)0.f;
  __syncthreads();
  for (int i = tid; i < NPIX * 64; i += 256) {
    const int p = i >> 6, c = i & 63;
    h_lds[((p / kW + 1) * HP + (p % kW + 1)) * 64 + c] = h0[(size_t)b * NPIX * 64 + i];
    c_lds[i] = c0[(size_t)b * NPIX * 64 + i];
  }
  bias_s[tid] = biasg[tid];
  __syncthreads();

  // per-lane pixel base offsets (in halves) for each M tile: invariant
  // across iterations, N tiles and K chunks — hoisted out of everything.
  const int koffL = (lane < 16) ? 0 : 8;
  int pixbase[MT];
#pragma unroll
  for (int mtt = 0; mtt < MT; ++mtt) {
    int row = mtt * 16 + (lane & 15);
    int p = row < NPIX ? row : NPIX - 1;          // clamp padded rows
    int py = p / kW, px = p - py * kW;
    pixbase[mtt] = (py * HP + px) * 64 + koffL;
  }

#pragma clang loop unroll(disable)
  for (int it = 0; it < NITER; ++it) {
    // ---- GEMM phase: gates = Wg (*) h ---------------------------------
#pragma clang loop unroll(disable)
    for (int nn = 0; nn < 2; ++nn) {
      const int nt = wave * 2 + nn;
      v8f acc[MT];
      v8f zero = {};
#pragma unroll
      for (int mtt = 0; mtt < MT; ++mtt) acc[mtt] = zero;

      // one K-chunk: 11 WMMAs with A double-buffered against LDS loads
      auto do_chunk = [&](int kc, const Frag& fb) {
        const int tap = kc >> 1, hc = kc & 1;
        const int delta = ((tap / 3) * HP + (tap - (tap / 3) * 3)) * 64 + hc * 32;
        Frag fa0, fa1;
        {
          const _Float16* s0 = h_lds + pixbase[0] + delta;
          fa0.u[0] = *(const u32x4*)s0;
          fa0.u[1] = *(const u32x4*)(s0 + 16);
        }
#pragma unroll
        for (int mtt = 0; mtt < MT; ++mtt) {
          if (mtt + 1 < MT) {                       // prefetch next A tile
            const _Float16* sn = h_lds + pixbase[mtt + 1] + delta;
            if (mtt & 1) { fa0.u[0] = *(const u32x4*)sn; fa0.u[1] = *(const u32x4*)(sn + 16); }
            else         { fa1.u[0] = *(const u32x4*)sn; fa1.u[1] = *(const u32x4*)(sn + 16); }
          }
          const Frag& cur = (mtt & 1) ? fa1 : fa0;
          acc[mtt] = __builtin_amdgcn_wmma_f32_16x16x32_f16(
              false, cur.h, false, fb.h, (short)0, acc[mtt], false, false);
        }
      };

      Frag fbA, fbB;
      {
        const u32x4* wp = wg + ((size_t)nt * 32 + lane) * 2;   // kc = 0
        fbA.u[0] = wp[0]; fbA.u[1] = wp[1];
      }
#pragma clang loop unroll(disable)
      for (int kk = 0; kk < KCH; kk += 2) {
        {
          const u32x4* wp = wg + ((size_t)((kk + 1) * NT + nt) * 32 + lane) * 2;
          fbB.u[0] = wp[0]; fbB.u[1] = wp[1];
          if (kk + 3 < KCH)                          // warm next pair in WGP$
            __builtin_prefetch((const void*)(wg + ((size_t)((kk + 3) * NT + nt) * 32 + lane) * 2), 0, 3);
        }
        do_chunk(kk, fbA);
        if (kk + 2 < KCH) {
          const u32x4* wp = wg + ((size_t)((kk + 2) * NT + nt) * 32 + lane) * 2;
          fbA.u[0] = wp[0]; fbA.u[1] = wp[1];
        }
        do_chunk(kk + 1, fbB);
      }

      // store D tiles to gate staging (f16)
      const int n = lane & 15;
      const int mofs = (lane < 16) ? 0 : 8;
#pragma unroll
      for (int mtt = 0; mtt < MT; ++mtt)
#pragma unroll
        for (int vv = 0; vv < 8; ++vv)
          g_lds[(mtt * 16 + mofs + vv) * NGATE + nt * 16 + n] = (_Float16)acc[mtt][vv];
    }
    __syncthreads();

    // ---- pointwise LSTM update ----------------------------------------
    for (int e = tid; e < NPIX * 64; e += 256) {
      const int p = e >> 6, c = e & 63;
      const float gi = (float)g_lds[p * NGATE + c]        + bias_s[c];
      const float gf = (float)g_lds[p * NGATE + 64 + c]   + bias_s[64 + c];
      const float gg = (float)g_lds[p * NGATE + 128 + c]  + bias_s[128 + c];
      const float go = (float)g_lds[p * NGATE + 192 + c]  + bias_s[192 + c];
      const float cc = sigmoidf_(gf) * c_lds[e] + sigmoidf_(gi) * tanhf(gg);
      c_lds[e] = cc;
      h_lds[((p / kW + 1) * HP + (p % kW + 1)) * 64 + c] = (_Float16)(sigmoidf_(go) * tanhf(cc));
    }
    __syncthreads();
  }

  for (int i = tid; i < NPIX * 64; i += 256) {
    const int p = i >> 6, c = i & 63;
    hout[(size_t)b * NPIX * 64 + i] = h_lds[((p / kW + 1) * HP + (p % kW + 1)) * 64 + c];
  }
}

// ---------------------------------------------------------------------------
// Readout: feat[g] = sum_{cap,p} h[p][g*8+cap]*satt[cap][p]; MLP 8->128->64->3
// ---------------------------------------------------------------------------
__global__ void __launch_bounds__(256)
readout_kernel(const _Float16* __restrict__ hf,   // [B][169][64]
               const float* __restrict__ satt,    // [B][8][169]
               const float* __restrict__ w1, const float* __restrict__ pb1,
               const float* __restrict__ w2, const float* __restrict__ pb2,
               const float* __restrict__ w3, const float* __restrict__ pb3,
               float* __restrict__ out)           // [B][3]
{
  __shared__ float red[256];
  __shared__ float feat[8];
  __shared__ float y1[128];
  __shared__ float y2[64];

  const int b = blockIdx.x, tid = threadIdx.x;
  const int lane = tid & 31, wv = tid >> 5;       // wave wv handles g = wv
  const _Float16* hb = hf + (size_t)b * NPIX * 64;
  const float* sb = satt + (size_t)b * CAPS * NPIX;

  float part = 0.f;
  for (int idx = lane; idx < CAPS * NPIX; idx += 32) {
    const int cap = idx / NPIX, p = idx - cap * NPIX;
    part += (float)hb[p * 64 + wv * CAPS + cap] * sb[cap * NPIX + p];
  }
  red[tid] = part;
  __syncthreads();
  for (int s = 16; s > 0; s >>= 1) {
    if (lane < s) red[tid] += red[tid + s];
    __syncthreads();
  }
  if (lane == 0) feat[wv] = red[tid];
  __syncthreads();

  if (tid < 128) {
    float a = pb1[tid];
#pragma unroll
    for (int k = 0; k < 8; ++k) a += w1[tid * 8 + k] * feat[k];
    y1[tid] = fmaxf(a, 0.f);
  }
  __syncthreads();
  if (tid < 64) {
    float a = pb2[tid];
    for (int k = 0; k < 128; ++k) a += w2[tid * 128 + k] * y1[k];
    y2[tid] = fmaxf(a, 0.f);
  }
  __syncthreads();
  if (tid < 3) {
    float a = pb3[tid];
    for (int k = 0; k < 64; ++k) a += w3[tid * 64 + k] * y2[k];
    out[b * 3 + tid] = a;
  }
}

// ---------------------------------------------------------------------------
// Launch. Param pointers assume jax-pytree flattening of the setup dict:
// top-level sorted keys [cur_state, goal_state, maze_map, params]; within
// params sorted keys, att_share list as (w,b)*6 in order.
// ---------------------------------------------------------------------------
extern "C" void kernel_launch(void* const* d_in, const int* in_sizes, int n_in,
                              void* d_out, int out_size, void* d_ws, size_t ws_size,
                              hipStream_t stream)
{
  (void)in_sizes; (void)out_size; (void)ws_size;
  if (n_in < 51) return;

  const float* cur_state  = (const float*)d_in[0];
  const float* goal_state = (const float*)d_in[1];
  const float* maze_map   = (const float*)d_in[2];
  const float* att_m_b1   = (const float*)d_in[3];
  const float* att_m_b2   = (const float*)d_in[4];
  const float* att_m_w1   = (const float*)d_in[5];
  const float* att_m_w2   = (const float*)d_in[6];
  const float* as_w0 = (const float*)d_in[7];
  const float* as_b0 = (const float*)d_in[8];
  const float* as_w1 = (const float*)d_in[9];
  const float* as_b1 = (const float*)d_in[10];
  const float* as_w2 = (const float*)d_in[11];
  const float* as_b2 = (const float*)d_in[12];
  const float* as_w3 = (const float*)d_in[13];
  const float* as_b3 = (const float*)d_in[14];
  const float* as_w4 = (const float*)d_in[15];
  const float* as_b4 = (const float*)d_in[16];
  const float* as_w5 = (const float*)d_in[17];
  const float* as_b5 = (const float*)d_in[18];
  const float* bn2_b = (const float*)d_in[19];
  const float* bn2_g = (const float*)d_in[20];
  const float* bn2_m = (const float*)d_in[21];
  const float* bn2_v = (const float*)d_in[22];
  const float* bn3_b = (const float*)d_in[23];
  const float* bn3_g = (const float*)d_in[24];
  const float* bn3_m = (const float*)d_in[25];
  const float* bn3_v = (const float*)d_in[26];
  const float* c0_b  = (const float*)d_in[27];
  const float* c0_w  = (const float*)d_in[28];
  const float* conv_b = (const float*)d_in[29];
  const float* conv_w = (const float*)d_in[30];
  const float* h0_b  = (const float*)d_in[31];
  const float* h0_w  = (const float*)d_in[32];
  const float* hid_b = (const float*)d_in[33];
  const float* hid_w = (const float*)d_in[34];
  const float* lstm_b_hh = (const float*)d_in[35];
  const float* lstm_b_ih = (const float*)d_in[36];
  const float* lstm_w_hh = (const float*)d_in[37];
  const float* lstm_w_ih = (const float*)d_in[38];
  const float* mc_b1 = (const float*)d_in[39];
  const float* mc_b2 = (const float*)d_in[40];
  const float* mc_b3 = (const float*)d_in[41];
  const float* mc_w1 = (const float*)d_in[42];
  const float* mc_w2 = (const float*)d_in[43];
  const float* mc_w3 = (const float*)d_in[44];
  const float* p_b1  = (const float*)d_in[45];
  const float* p_b2  = (const float*)d_in[46];
  const float* p_b3  = (const float*)d_in[47];
  const float* p_w1  = (const float*)d_in[48];
  const float* p_w2  = (const float*)d_in[49];
  const float* p_w3  = (const float*)d_in[50];

  // workspace layout (bytes); regions reused once their producer is consumed
  char* ws = (char*)d_ws;
  float*    gatt   = (float*)(ws + 0);               // 512*8*169*4   = 2,768,896
  float*    satt   = (float*)(ws + 2768896);         // 2,768,896
  float*    biasg  = (float*)(ws + 5537792);         // 1,024
  unsigned* wgfrag = (unsigned*)(ws + 5538816);      // 294,912
  float*    c1out  = (float*)(ws + 5833728);         // 512*16*2500*4 = 81,920,000
  float*    c2out  = (float*)(ws + 87753728);        // 512*32*625*4  = 40,960,000
  float*    mfeat  = (float*)(ws + 128713728);       // 512*64*169*4  = 22,151,168
  float*    hlayer = (float*)(ws + 5833728);         // reuses c1out region
  _Float16* h0buf  = (_Float16*)(ws + 87753728);     // reuses c2out region (11,075,584)
  float*    c0buf  = (float*)(ws + 98829312);        // 22,151,168 (still inside c2out+)
  _Float16* hfin   = (_Float16*)(ws + 128713728);    // reuses mfeat region

  AttnW aw = { as_w0, as_b0, as_w1, as_b1, as_w2, as_b2, as_w3, as_b3,
               as_w4, as_b4, as_w5, as_b5, att_m_w1, att_m_b1, att_m_w2, att_m_b2 };

  attn_kernel<<<dim3(kB, 2), 192, 0, stream>>>(goal_state, cur_state, aw, gatt, satt);
  prep_kernel<<<KCH * NT, 256, 0, stream>>>(conv_w, lstm_w_ih, lstm_w_hh,
                                            lstm_b_ih, lstm_b_hh, conv_b, wgfrag, biasg);
  conv1_kernel<<<(kB * 16 * 2500 + 255) / 256, 256, 0, stream>>>(maze_map, mc_w1, mc_b1, c1out);
  conv2_kernel<<<(kB * 32 * 625 + 255) / 256, 256, 0, stream>>>(c1out, mc_w2, mc_b2,
                                                                bn2_g, bn2_b, bn2_m, bn2_v, c2out);
  conv3_kernel<<<(kB * 64 * NPIX + 255) / 256, 256, 0, stream>>>(c2out, mc_w3, mc_b3,
                                                                 bn3_g, bn3_b, bn3_m, bn3_v, mfeat);
  hid_kernel<<<(kB * 64 * NPIX + 255) / 256, 256, 0, stream>>>(mfeat, gatt, hid_w, hid_b, hlayer);
  h0c0_kernel<<<(kB * 64 * NPIX + 255) / 256, 256, 0, stream>>>(hlayer, h0_w, h0_b,
                                                                c0_w, c0_b, h0buf, c0buf);
  lstm_kernel<<<kB, 256, 0, stream>>>(h0buf, c0buf, (const u32x4*)wgfrag, biasg, hfin);
  readout_kernel<<<kB, 256, 0, stream>>>(hfin, satt, p_w1, p_b1, p_w2, p_b2, p_w3, p_b3,
                                         (float*)d_out);
}
